// process_video_sequence_8443905704305
// MI455X (gfx1250) — compile-verified, matrix operand order checked by reading, not run on hardware
//
#include <hip/hip_runtime.h>
#include <hip/hip_bf16.h>

// ---------------------------------------------------------------------------
// MI455X (gfx1250) implementation — see round-0 analysis for the algebraic
// restructuring (parallel GEMM phase + degenerate text-side attention +
// 128-step serial recurrence).
//
// This round: tile staging switched from VGPR double-buffering (which spilled
// to scratch) to gfx1250 async copies: global_load_async_to_lds_b128 writes
// LDS directly (no data VGPRs), tracked by ASYNCcnt; compute on buffer k
// overlaps the async DMA of buffer k+1, synchronized by
// s_wait_asynccnt 0 + workgroup barrier.
// ---------------------------------------------------------------------------

typedef __bf16 bf16;
typedef __attribute__((ext_vector_type(16))) __bf16 v16bf;
typedef __attribute__((ext_vector_type(8)))  float  v8f;

static constexpr int IN_DIM  = 768;
static constexpr int HID     = 1024;
static constexpr int OUT_DIM = 768;
static constexpr int BATCH   = 256;
static constexpr int NFRAMES = 128;
static constexpr int BT      = BATCH * NFRAMES;   // 32768

#define BKT 32   // K-tile (one WMMA K per stage)

// Async global->LDS copy, 16B per lane. VDST = LDS byte address, VADDR = 64b
// global address (GV mode). Tracked by ASYNCcnt.
__device__ __forceinline__ void async_copy_b128(unsigned lds_off, const void* gptr) {
    asm volatile("global_load_async_to_lds_b128 %0, %1, off"
                 :: "v"(lds_off), "v"((unsigned long long)(uintptr_t)gptr)
                 : "memory");
}
__device__ __forceinline__ void wait_async_all() {
    asm volatile("s_wait_asynccnt 0x0" ::: "memory");
}

// ---------------------------------------------------------------------------
// WMMA GEMM: C[M,N] = A[M,K](bf16,rm) * W[N,K](bf16,rm)^T + bias + addend
// addendMode: 0 none | 1 bf16 row=m | 2 bf16 row=(m % BATCH)
// permuteBT : out row m = t*BATCH+b written at row (b*NFRAMES + t)
// 256 threads = 8 wave32 waves in a 2(M) x 4(N) grid; each wave computes a
// (WM_T*16) x (WN_T*16) patch with WM_T*WN_T WMMA tiles.
//   big:   WM_T=4, WN_T=4  -> block 128 x 256
//   small: WM_T=2, WN_T=2  -> block  64 x 128
// ---------------------------------------------------------------------------
template <int WM_T, int WN_T>
__global__ __launch_bounds__(256)
void gemm_bf16_wmma_k(const bf16* __restrict__ A, int lda,
                      const bf16* __restrict__ W, int ldw,
                      const float* __restrict__ bias,
                      const bf16* __restrict__ addend, int addendMode, int ldadd,
                      float* __restrict__ outF, bf16* __restrict__ outB,
                      int M, int N, int K, int permuteBT)
{
    constexpr int BM = 2 * 16 * WM_T;
    constexpr int BN = 4 * 16 * WN_T;

    __shared__ bf16 sA[2][BM * BKT];
    __shared__ bf16 sB[2][BN * BKT];

    const int tid    = threadIdx.x;
    const int lane   = tid & 31;
    const int wave   = tid >> 5;
    const int waveM  = wave & 1;
    const int waveN  = wave >> 1;
    const int blockM = blockIdx.x * BM;
    const int blockN = blockIdx.y * BN;

    const int half = lane >> 4;     // 16-bit A/B frag layout: lane<16 -> K{0..7,16..23}
    const int lrow = lane & 15;     //                         lane>=16 -> K{8..15,24..31}

    v8f acc[WM_T][WN_T] = {};

    const int nK = K / BKT;

    auto stageAsync = [&](int kb, int buf) {
        const int k0 = kb * BKT;
#pragma unroll
        for (int i = tid; i < BM * 4; i += 256) {          // BM*32 bf16 / 8 per b128
            const int row = i >> 2, kc = (i & 3) * 8;
            async_copy_b128((unsigned)(uintptr_t)&sA[buf][row * BKT + kc],
                            &A[(size_t)(blockM + row) * lda + k0 + kc]);
        }
#pragma unroll
        for (int i = tid; i < BN * 4; i += 256) {
            const int row = i >> 2, kc = (i & 3) * 8;
            async_copy_b128((unsigned)(uintptr_t)&sB[buf][row * BKT + kc],
                            &W[(size_t)(blockN + row) * ldw + k0 + kc]);
        }
    };

    stageAsync(0, 0);
    wait_async_all();
    __syncthreads();

    for (int kb = 0; kb < nK; ++kb) {
        const int buf = kb & 1;
        if (kb + 1 < nK) stageAsync(kb + 1, buf ^ 1);   // DMA next tile, no VGPR data

        // A fragments for this wave's rows
        v16bf af[WM_T];
#pragma unroll
        for (int mt = 0; mt < WM_T; ++mt) {
            const int r = waveM * (WM_T * 16) + mt * 16 + lrow;
            const bf16* p = &sA[buf][r * BKT + half * 8];
            union { v16bf v; uint4 u[2]; } t;
            t.u[0] = *(const uint4*)p;
            t.u[1] = *(const uint4*)(p + 16);
            af[mt] = t.v;
        }
        // B fragments one at a time (keeps live VGPRs ~ acc + af + 1 frag)
#pragma unroll
        for (int nt = 0; nt < WN_T; ++nt) {
            const int r = waveN * (WN_T * 16) + nt * 16 + lrow;
            const bf16* p = &sB[buf][r * BKT + half * 8];
            union { v16bf v; uint4 u[2]; } t;
            t.u[0] = *(const uint4*)p;
            t.u[1] = *(const uint4*)(p + 16);
            const v16bf bg = t.v;
#pragma unroll
            for (int mt = 0; mt < WM_T; ++mt)
                acc[mt][nt] = __builtin_amdgcn_wmma_f32_16x16x32_bf16(
                    false, af[mt], false, bg, (short)0, acc[mt][nt],
                    false, false);
        }

        wait_async_all();       // our async writes into buf^1 have landed
        __syncthreads();        // everyone's have
    }

    // Epilogue. C layout: VGPR g -> M=g (lanes 0-15) / M=g+8 (lanes 16-31), N=lane%16.
    const int rowOff = (lane >> 4) * 8;
    const int col    = lane & 15;
#pragma unroll
    for (int mt = 0; mt < WM_T; ++mt) {
#pragma unroll
        for (int nt = 0; nt < WN_T; ++nt) {
            const int n = blockN + waveN * (WN_T * 16) + nt * 16 + col;
            const float bia = bias ? bias[n] : 0.f;
#pragma unroll
            for (int g = 0; g < 8; ++g) {
                const int m = blockM + waveM * (WM_T * 16) + mt * 16 + g + rowOff;
                float v = acc[mt][nt][g] + bia;
                if (addendMode == 1)
                    v += (float)addend[(size_t)m * ldadd + n];
                else if (addendMode == 2)
                    v += (float)addend[(size_t)(m & (BATCH - 1)) * ldadd + n];
                size_t o;
                if (permuteBT) {
                    const int t = m >> 8;            // m = t*BATCH + b
                    const int b = m & (BATCH - 1);
                    o = ((size_t)b * NFRAMES + t) * (size_t)N + n;
                } else {
                    o = (size_t)m * (size_t)N + n;
                }
                if (outF) outF[o] = v;
                if (outB) outB[o] = (bf16)v;
            }
        }
    }
}

// ---------------------------------------------------------------------------
// Sliding-window cross-attention (text query, video keys/values).
// ---------------------------------------------------------------------------
__global__ __launch_bounds__(128)
void attn_window_k(const bf16* __restrict__ qT,   // (256,1024)
                   const bf16* __restrict__ kF,   // (T*B,1024) time-major
                   const bf16* __restrict__ vF,   // (T*B,1024) time-major
                   bf16* __restrict__ ctx)        // (T*B,1024) time-major
{
    const int bt = blockIdx.x;
    const int t  = bt >> 8;
    const int b  = bt & (BATCH - 1);
    const int t0 = (t > 12) ? (t - 12) : 0;
    const int L  = t - t0 + 1;

    __shared__ float w[4][16];
    const int tid = threadIdx.x;

    if (tid < 4 * 13) {
        const int h = tid / 13, i = tid % 13;
        float s = -1e30f;
        if (i < L) {
            const bf16* q = qT + (size_t)b * HID + h * 256;
            const bf16* k = kF + ((size_t)(t0 + i) * BATCH + b) * HID + h * 256;
            float a = 0.f;
            for (int d = 0; d < 256; ++d) a += (float)q[d] * (float)k[d];
            s = a * 0.0625f;  // 1/sqrt(256)
        }
        w[h][i] = s;
    }
    __syncthreads();
    if (tid < 4) {
        const int h = tid;
        float mx = -1e30f;
        for (int i = 0; i < L; ++i) mx = fmaxf(mx, w[h][i]);
        float sum = 0.f;
        for (int i = 0; i < L; ++i) { float e = __expf(w[h][i] - mx); w[h][i] = e; sum += e; }
        const float inv = 1.f / sum;
        for (int i = 0; i < L; ++i) w[h][i] *= inv;
    }
    __syncthreads();
    for (int d = tid; d < HID; d += 128) {
        const int h = d >> 8;
        float a = 0.f;
        for (int i = 0; i < L; ++i)
            a += w[h][i] * (float)vF[((size_t)(t0 + i) * BATCH + b) * HID + d];
        ctx[(size_t)bt * HID + d] = (bf16)a;
    }
}

// ---------------------------------------------------------------------------
// Conversion / init kernels
// ---------------------------------------------------------------------------
__global__ void cvt_f32_bf16_k(const float* __restrict__ in, bf16* __restrict__ out, int n) {
    const int i = blockIdx.x * blockDim.x + threadIdx.x;
    if (i < n) out[i] = (bf16)in[i];
}

__global__ void frames_to_tm_k(const float* __restrict__ in, bf16* __restrict__ out) {
    // (B,T,768) f32 -> (T,B,768) bf16
    const int i = blockIdx.x * blockDim.x + threadIdx.x;
    if (i < BT * IN_DIM) {
        const int c  = i % IN_DIM;
        const int bt = i / IN_DIM;
        const int t  = bt % NFRAMES;
        const int b  = bt / NFRAMES;
        out[((size_t)t * BATCH + b) * IN_DIM + c] = (bf16)in[i];
    }
}

__global__ void zero_bf16_k(bf16* p, int n) {
    const int i = blockIdx.x * blockDim.x + threadIdx.x;
    if (i < n) p[i] = (bf16)0.f;
}

// ---------------------------------------------------------------------------
extern "C" void kernel_launch(void* const* d_in, const int* in_sizes, int n_in,
                              void* d_out, int out_size, void* d_ws, size_t ws_size,
                              hipStream_t stream) {
    const float* f_frames  = (const float*)d_in[0];
    const float* f_text    = (const float*)d_in[1];
    const float* f_conv_w  = (const float*)d_in[2];
    const float* b_conv    = (const float*)d_in[3];
    const float* f_Av_w    = (const float*)d_in[4];
    const float* b_Av      = (const float*)d_in[5];
    const float* f_Bv_w    = (const float*)d_in[6];
    const float* b_Bv      = (const float*)d_in[7];
    const float* f_Cv_w    = (const float*)d_in[8];
    const float* b_Cv      = (const float*)d_in[9];
    const float* f_Dv_w    = (const float*)d_in[10];
    const float* b_Dv      = (const float*)d_in[11];
    const float* f_Ac_w    = (const float*)d_in[12];
    const float* b_Ac      = (const float*)d_in[13];
    const float* f_Bc_w    = (const float*)d_in[14];
    const float* b_Bc      = (const float*)d_in[15];
    const float* f_Cc_w    = (const float*)d_in[16];
    const float* b_Cc      = (const float*)d_in[17];
    const float* f_Dc_w    = (const float*)d_in[18];
    const float* b_Dc      = (const float*)d_in[19];
    const float* f_qkv_w   = (const float*)d_in[20];
    const float* b_qkv     = (const float*)d_in[21];
    const float* f_out_w   = (const float*)d_in[22];
    const float* b_outp    = (const float*)d_in[23];

    size_t off = 0;
    auto alloc = [&](size_t bytes) -> char* {
        char* p = (char*)d_ws + off;
        off += (bytes + 255) & ~(size_t)255;
        return p;
    };
    bf16* framesTM = (bf16*)alloc((size_t)BT * IN_DIM * 2);
    bf16* textB    = (bf16*)alloc((size_t)BATCH * IN_DIM * 2);
    bf16* wConv    = (bf16*)alloc((size_t)HID * IN_DIM * 2);
    bf16* wQKV     = (bf16*)alloc((size_t)3 * HID * HID * 2);
    bf16* wOut     = (bf16*)alloc((size_t)HID * HID * 2);
    bf16* wAv      = (bf16*)alloc((size_t)HID * HID * 2);
    bf16* wBv      = (bf16*)alloc((size_t)HID * HID * 2);
    bf16* wAc      = (bf16*)alloc((size_t)HID * HID * 2);
    bf16* wBc      = (bf16*)alloc((size_t)HID * HID * 2);
    bf16* wCv      = (bf16*)alloc((size_t)OUT_DIM * HID * 2);
    bf16* wCc      = (bf16*)alloc((size_t)OUT_DIM * HID * 2);
    bf16* wDv      = (bf16*)alloc((size_t)OUT_DIM * IN_DIM * 2);
    bf16* wDc      = (bf16*)alloc((size_t)OUT_DIM * IN_DIM * 2);
    bf16* slotA    = (bf16*)alloc((size_t)BT * HID * 2);   // hF -> ctx -> hAllC
    bf16* slotB    = (bf16*)alloc((size_t)BT * HID * 2);   // kF -> bcAll
    bf16* slotC    = (bf16*)alloc((size_t)BT * HID * 2);   // vF -> hAllV
    bf16* slotD    = (bf16*)alloc((size_t)BT * HID * 2);   // hcCur -> dv
    bf16* hT       = (bf16*)alloc((size_t)BATCH * HID * 2);
    bf16* qT       = (bf16*)alloc((size_t)BATCH * HID * 2);
    bf16* vT       = (bf16*)alloc((size_t)BATCH * HID * 2);
    bf16* hvCur    = (bf16*)alloc((size_t)BATCH * HID * 2);
    bf16* bvBuf    = (bf16*)alloc((size_t)BATCH * HID * 2);
    bf16* dcBuf    = (bf16*)alloc((size_t)BATCH * OUT_DIM * 2);
    bf16* zeroB    = (bf16*)alloc((size_t)BATCH * HID * 2);

    bf16* hF    = slotA; bf16* ctx   = slotA; bf16* hAllC = slotA;
    bf16* kF    = slotB; bf16* bcAll = slotB;
    bf16* vF    = slotC; bf16* hAllV = slotC;
    bf16* hcCur = slotD; bf16* dv    = slotD;

    const bf16* wWq = wQKV;
    const bf16* wWk = wQKV + (size_t)HID * HID;
    const bf16* wWv = wQKV + (size_t)2 * HID * HID;
    const float* bq  = b_qkv;
    const float* bk  = b_qkv + HID;
    const float* bvv = b_qkv + 2 * HID;

    auto cvt = [&](const float* in, bf16* out, int n) {
        cvt_f32_bf16_k<<<dim3((n + 255) / 256), dim3(256), 0, stream>>>(in, out, n);
    };
    // big tiles (128x256) for the parallel phase
    auto gemm = [&](const bf16* A, int lda, const bf16* W, int ldw,
                    const float* bias, const bf16* add, int addMode, int ldadd,
                    float* oF, bf16* oB, int M, int N, int K, int perm) {
        dim3 g(M / 128, N / 256);
        gemm_bf16_wmma_k<4, 4><<<g, dim3(256), 0, stream>>>(
            A, lda, W, ldw, bias, add, addMode, ldadd, oF, oB, M, N, K, perm);
    };
    // small tiles (64x128) for the latency-critical serial recurrence
    auto gemmS = [&](const bf16* A, const bf16* W, const float* bias,
                     const bf16* add, bf16* oB) {
        dim3 g(BATCH / 64, HID / 128);
        gemm_bf16_wmma_k<2, 2><<<g, dim3(256), 0, stream>>>(
            A, HID, W, HID, bias, add, 1, HID, nullptr, oB, BATCH, HID, HID, 0);
    };

    // ---- phase 0: precision conversion (bf16 weights, ~24MB, L2-resident)
    frames_to_tm_k<<<dim3((BT * IN_DIM + 255) / 256), dim3(256), 0, stream>>>(f_frames, framesTM);
    cvt(f_text,   textB, BATCH * IN_DIM);
    cvt(f_conv_w, wConv, HID * IN_DIM);
    cvt(f_qkv_w,  wQKV,  3 * HID * HID);
    cvt(f_out_w,  wOut,  HID * HID);
    cvt(f_Av_w,   wAv,   HID * HID);
    cvt(f_Bv_w,   wBv,   HID * HID);
    cvt(f_Ac_w,   wAc,   HID * HID);
    cvt(f_Bc_w,   wBc,   HID * HID);
    cvt(f_Cv_w,   wCv,   OUT_DIM * HID);
    cvt(f_Cc_w,   wCc,   OUT_DIM * HID);
    cvt(f_Dv_w,   wDv,   OUT_DIM * IN_DIM);
    cvt(f_Dc_w,   wDc,   OUT_DIM * IN_DIM);
    zero_bf16_k<<<dim3((BATCH * HID + 255) / 256), dim3(256), 0, stream>>>(zeroB, BATCH * HID);

    // ---- phase 1: parallel GEMMs over all timesteps
    gemm(framesTM, IN_DIM, wConv, IN_DIM, b_conv, nullptr, 0, 0, nullptr, hF, BT,    HID, IN_DIM, 0);
    gemm(textB,    IN_DIM, wConv, IN_DIM, b_conv, nullptr, 0, 0, nullptr, hT, BATCH, HID, IN_DIM, 0);
    gemm(hF, HID, wWk, HID, bk,  nullptr, 0, 0, nullptr, kF, BT,    HID, HID, 0);
    gemm(hF, HID, wWv, HID, bvv, nullptr, 0, 0, nullptr, vF, BT,    HID, HID, 0);
    gemm(hT, HID, wWq, HID, bq,  nullptr, 0, 0, nullptr, qT, BATCH, HID, HID, 0);
    gemm(hT, HID, wWv, HID, bvv, nullptr, 0, 0, nullptr, vT, BATCH, HID, HID, 0);
    // video-path attention is degenerate (all text keys identical) -> out = vT proj
    gemm(vT, HID, wOut, HID, b_outp, nullptr, 0, 0, nullptr, hvCur, BATCH, HID, HID, 0);

    // ---- phase 2: window attention (text query, video keys), then out-proj
    attn_window_k<<<dim3(BT), dim3(128), 0, stream>>>(qT, kF, vF, ctx);
    gemm(ctx, HID, wOut, HID, b_outp, nullptr, 0, 0, nullptr, hcCur, BT, HID, HID, 0);

    // ---- phase 3: recurrence addends (parallel over t)
    gemm(hvCur, HID, wBv, HID, b_Bv, nullptr, 0, 0, nullptr, bvBuf, BATCH, HID, HID, 0);
    gemm(hcCur, HID, wBc, HID, b_Bc, nullptr, 0, 0, nullptr, bcAll, BT,    HID, HID, 0);
    gemm(framesTM, IN_DIM, wDv, IN_DIM, b_Dv, nullptr, 0, 0, nullptr, dv,    BT,    OUT_DIM, IN_DIM, 0);
    gemm(textB,    IN_DIM, wDc, IN_DIM, b_Dc, nullptr, 0, 0, nullptr, dcBuf, BATCH, OUT_DIM, IN_DIM, 0);

    // ---- phase 4: serial scan  h[t] = h[t-1] @ A^T + A_b + addend[t]
    const size_t S = (size_t)BATCH * HID;
    for (int t = 0; t < NFRAMES; ++t) {
        const bf16* pv = (t == 0) ? zeroB : (hAllV + (size_t)(t - 1) * S);
        const bf16* pc = (t == 0) ? zeroB : (hAllC + (size_t)(t - 1) * S);
        gemmS(pv, wAv, b_Av, bvBuf,                 hAllV + (size_t)t * S);
        gemmS(pc, wAc, b_Ac, bcAll + (size_t)t * S, hAllC + (size_t)t * S);
    }

    // ---- phase 5: outputs, permuted back to (B,T,OUT)
    float* outYv = (float*)d_out;
    float* outYc = (float*)d_out + (size_t)BT * OUT_DIM;
    gemm(hAllV, HID, wCv, HID, b_Cv, dv,    1, OUT_DIM, outYv, nullptr, BT, OUT_DIM, HID, 1);
    gemm(hAllC, HID, wCc, HID, b_Cc, dcBuf, 2, OUT_DIM, outYc, nullptr, BT, OUT_DIM, HID, 1);
}
